// LDGCNNSegmentor_2731599200340
// MI455X (gfx1250) — matrix operand order repaired and checked by reading, not compiled, not run on hardware
//
#include <hip/hip_runtime.h>
#include <hip/hip_bf16.h>
#include <cstdint>

// ---------------------------------------------------------------------------
// LDGCNN segmentor for MI455X (gfx1250, wave32, WMMA).
// Dense layers: v_wmma_f32_16x16x32_bf16 (bf16 in, f32 accumulate).
// Fragment feeding: 2x b128 loads per fragment (LDS for A, swizzled global
// for B). Head GEMM A-tiles staged via the Tensor Data Mover when available.
// KNN is exact f32 (VALU) with LDS candidate tiling + LDS top-k lists.
// Workspace use: ~200 MB.
// ---------------------------------------------------------------------------

typedef __attribute__((ext_vector_type(16))) __bf16 v16bf;
typedef __attribute__((ext_vector_type(8)))  __bf16 v8bf;
typedef __attribute__((ext_vector_type(8)))  float  v8f;

static constexpr int BATCH = 8;
static constexpr int NPTS  = 4096;
static constexpr int KNN   = 30;
static constexpr int MROWS = BATCH * NPTS;   // 32768 flattened rows

#if __has_builtin(__builtin_amdgcn_tensor_load_to_lds)
#define HAVE_TDM 1
typedef unsigned int u32x4 __attribute__((ext_vector_type(4)));
typedef int          i32x4 __attribute__((ext_vector_type(4)));
typedef int          i32x8 __attribute__((ext_vector_type(8)));
#endif

// ---- WMMA helpers ----------------------------------------------------------

__device__ __forceinline__ v8f wmma_bf16(v16bf a, v16bf b, v8f c) {
  return __builtin_amdgcn_wmma_f32_16x16x32_bf16(false, a, false, b,
                                                 (short)0, c, false, false);
}

// A fragment: 16x32 bf16 from a row-major tile (LDS), leading dim = ld (elems).
// Lane (row = lane%16, hl = lane/16) holds K = {hl*8..+7, 16+hl*8..+7}:
// two contiguous 16-byte runs -> two ds_load_b128.
__device__ __forceinline__ v16bf ld_a_lds(const __bf16* base, int ld, int lane) {
  const __bf16* p = base + (lane & 15) * ld + ((lane >> 4) << 3);
  v8bf lo = *(const v8bf*)p;
  v8bf hi = *(const v8bf*)(p + 16);
  return __builtin_shufflevector(lo, hi, 0, 1, 2, 3, 4, 5, 6, 7,
                                 8, 9, 10, 11, 12, 13, 14, 15);
}

// B fragment from swizzled weights: per (ktile,ntile) the 32 lanes' 16 bf16
// values are contiguous (32 bytes per lane) -> one 32-byte vector load.
__device__ __forceinline__ v16bf ld_b_swz(const __bf16* __restrict__ Wsw,
                                          int ntiles, int kt, int nt, int lane) {
  const __bf16* p = Wsw + (((size_t)(kt * ntiles + nt) * 32 + lane) << 4);
  return *(const v16bf*)p;
}

// ---- small utility kernels -------------------------------------------------

__global__ void build_x0_kernel(const float* __restrict__ x,
                                const float* __restrict__ pos,
                                float* __restrict__ x0) {
  int i = blockIdx.x * 256 + threadIdx.x;
  if (i < MROWS * 6) {
    int row = i / 6, c = i % 6;
    x0[i] = (c < 3) ? x[(size_t)row * 3 + c] : pos[(size_t)row * 3 + (c - 3)];
  }
}

__global__ void f2bf_kernel(const float* __restrict__ s, __bf16* __restrict__ d,
                            int count) {
  int i = blockIdx.x * 256 + threadIdx.x;
  if (i < count) d[i] = (__bf16)s[i];
}

// Convert f32 weight (K x Ns) to bf16 fragment-swizzled (Kp x Np), zero pad.
// dst[((kt*ntiles + nt)*32 + lane)*16 + e] =
//   W[kt*32 + (lane/16)*8 + (e%8) + (e/8)*16][nt*16 + lane%16]
__global__ void wcvt_swz_kernel(const float* __restrict__ s, __bf16* __restrict__ d,
                                int K, int Ns, int Kp, int Np) {
  int i = blockIdx.x * 256 + threadIdx.x;
  int total = Kp * Np;
  if (i < total) {
    int ntiles = Np >> 4;
    int e    = i & 15;
    int lane = (i >> 4) & 31;
    int tile = i >> 9;
    int kt   = tile / ntiles;
    int nt   = tile - kt * ntiles;
    int col  = nt * 16 + (lane & 15);
    int k    = kt * 32 + ((lane >> 4) << 3) + (e & 7) + ((e >> 3) << 4);
    d[i] = (k < K && col < Ns) ? (__bf16)s[(size_t)k * Ns + col] : (__bf16)0.0f;
  }
}

// ---- KNN: exact f32 distances, LDS tiling + LDS top-k ----------------------

template <int C>
__global__ void __launch_bounds__(128) knn_kernel(const float* __restrict__ x,
                                                  int* __restrict__ idxo) {
  __shared__ float tile[128 * C];
  __shared__ float tnorm[128];
  __shared__ float bestd[KNN * 128];
  __shared__ int   besti[KNN * 128];
  int tid = threadIdx.x;
  int b   = blockIdx.y;
  int n   = blockIdx.x * 128 + tid;
  const float* xb = x + (size_t)b * NPTS * C;

  float xf[C];
  float xx = 0.f;
#pragma unroll
  for (int c = 0; c < C; ++c) {
    xf[c] = xb[(size_t)n * C + c];
    xx += xf[c] * xf[c];
  }
#pragma unroll
  for (int q = 0; q < KNN; ++q) {
    bestd[q * 128 + tid] = -3.0e38f;
    besti[q * 128 + tid] = 0;
  }
  for (int t = 0; t < NPTS / 128; ++t) {
    __syncthreads();
    for (int i = tid; i < 128 * C; i += 128)
      tile[i] = xb[(size_t)(t * 128) * C + i];
    __syncthreads();
    float s = 0.f;
#pragma unroll
    for (int c = 0; c < C; ++c) { float v = tile[tid * C + c]; s += v * v; }
    tnorm[tid] = s;
    __syncthreads();
    for (int j = 0; j < 128; ++j) {
      float dot = 0.f;
#pragma unroll
      for (int c = 0; c < C; ++c) dot += xf[c] * tile[j * C + c];
      float d = 2.f * dot - xx - tnorm[j];
      if (d > bestd[(KNN - 1) * 128 + tid]) {
        int m = t * 128 + j;
        int p = KNN - 1;
        while (p > 0 && bestd[(p - 1) * 128 + tid] < d) {
          bestd[p * 128 + tid] = bestd[(p - 1) * 128 + tid];
          besti[p * 128 + tid] = besti[(p - 1) * 128 + tid];
          --p;
        }
        bestd[p * 128 + tid] = d;
        besti[p * 128 + tid] = m;
      }
    }
  }
  int* o = idxo + ((size_t)b * NPTS + n) * KNN;
#pragma unroll
  for (int q = 0; q < KNN; ++q) o[q] = besti[q * 128 + tid];
}

// ---- EdgeConv: per-wave point; edge matrix in LDS; two WMMA GEMMs + max ----

template <int CIN, int KPAD, int H, int WAVES>
__global__ void __launch_bounds__(WAVES * 32) edgeconv_kernel(
    const float* __restrict__ x, const int* __restrict__ idx,
    const __bf16* __restrict__ W1, const float* __restrict__ b1,
    const __bf16* __restrict__ W2, const float* __restrict__ b2,
    float* __restrict__ out) {
  __shared__ alignas(32) __bf16 Eb[WAVES][32 * KPAD];  // edges (32 x KPAD)
  __shared__ alignas(32) __bf16 Hb[WAVES][32 * H];     // relu(E@W1+b1)
  int tid  = threadIdx.x;
  int wave = tid >> 5, lane = tid & 31;
  int b = blockIdx.y;
  int n = blockIdx.x * WAVES + wave;
  const float* xb = x + (size_t)b * NPTS * CIN;
  const int*   ib = idx + ((size_t)b * NPTS + n) * KNN;

  // Build E = [xi | nbr - xi] (rows 30,31 duplicate edge 0; cols >= 2*CIN zero)
  for (int i = lane; i < 32 * KPAD; i += 32) {
    int r = i / KPAD, c = i % KPAD;
    float v = 0.f;
    if (c < CIN) {
      v = xb[(size_t)n * CIN + c];
    } else if (c < 2 * CIN) {
      int e  = (r < KNN) ? r : 0;
      int m  = ib[e];
      int cc = c - CIN;
      v = xb[(size_t)m * CIN + cc] - xb[(size_t)n * CIN + cc];
    }
    Eb[wave][r * KPAD + c] = (__bf16)v;
  }
  __syncthreads();

  int colL = lane & 15, hl = lane >> 4;
  // GEMM1: (32 x KPAD) @ (KPAD x H) -> relu -> Hb
#pragma unroll
  for (int mt = 0; mt < 2; ++mt)
    for (int nt = 0; nt < H / 16; ++nt) {
      v8f acc = {};
#pragma unroll
      for (int ks = 0; ks < KPAD / 32; ++ks) {
        v16bf a  = ld_a_lds(&Eb[wave][mt * 16 * KPAD + ks * 32], KPAD, lane);
        v16bf bb = ld_b_swz(W1, H / 16, ks, nt, lane);
        acc = wmma_bf16(a, bb, acc);
      }
      int col    = nt * 16 + colL;
      float bias = b1[col];
#pragma unroll
      for (int r = 0; r < 8; ++r) {
        float v = acc[r] + bias;
        v = v > 0.f ? v : 0.f;
        Hb[wave][(mt * 16 + hl * 8 + r) * H + col] = (__bf16)v;
      }
    }
  __syncthreads();

  // GEMM2: (32 x H) @ (H x H) + b2, then max over edge rows
  for (int nt = 0; nt < H / 16; ++nt) {
    float vmax = -3.0e38f;
#pragma unroll
    for (int mt = 0; mt < 2; ++mt) {
      v8f acc = {};
#pragma unroll
      for (int ks = 0; ks < H / 32; ++ks) {
        v16bf a  = ld_a_lds(&Hb[wave][mt * 16 * H + ks * 32], H, lane);
        v16bf bb = ld_b_swz(W2, H / 16, ks, nt, lane);
        acc = wmma_bf16(a, bb, acc);
      }
#pragma unroll
      for (int r = 0; r < 8; ++r) vmax = fmaxf(vmax, acc[r]);
    }
    vmax = fmaxf(vmax, __shfl_xor(vmax, 16, 32));
    if (hl == 0)
      out[((size_t)b * NPTS + n) * H + nt * 16 + colL] = vmax + b2[nt * 16 + colL];
  }
}

// ---- Fused fe MLP (128->1024 relu ->1024) + per-block column max -----------

__global__ void __launch_bounds__(256) fe_fused_kernel(
    const __bf16* __restrict__ A, const __bf16* __restrict__ W1,
    const float* __restrict__ b1, const __bf16* __restrict__ W2,
    const float* __restrict__ b2, float* __restrict__ partials) {
  __shared__ alignas(32) __bf16 At[16 * 128];    // 4 KB
  __shared__ alignas(32) __bf16 Hb[16 * 1024];   // 32 KB
  int tid  = threadIdx.x;
  int wave = tid >> 5, lane = tid & 31;
  int colL = lane & 15, hl = lane >> 4;
  size_t row0   = (size_t)blockIdx.x * 16;
  int    batch  = (int)(row0 / NPTS);
  int    blkInB = (int)((row0 % NPTS) / 16);

  // stage A tile (16x128 bf16 = 4 KB) with 16B vector copies
  ((uint4*)At)[tid] = ((const uint4*)(A + row0 * 128))[tid];
  __syncthreads();

#pragma unroll
  for (int j = 0; j < 8; ++j) {
    int nt = wave * 8 + j;
    v8f acc = {};
#pragma unroll
    for (int ks = 0; ks < 4; ++ks) {
      v16bf a  = ld_a_lds(&At[ks * 32], 128, lane);
      v16bf bb = ld_b_swz(W1, 64, ks, nt, lane);
      acc = wmma_bf16(a, bb, acc);
    }
    int col    = nt * 16 + colL;
    float bias = b1[col];
#pragma unroll
    for (int r = 0; r < 8; ++r) {
      float v = acc[r] + bias;
      v = v > 0.f ? v : 0.f;
      Hb[(hl * 8 + r) * 1024 + col] = (__bf16)v;
    }
  }
  __syncthreads();

  for (int j = 0; j < 8; ++j) {
    int nt = wave * 8 + j;
    v8f acc = {};
    for (int ks = 0; ks < 32; ++ks) {
      v16bf a  = ld_a_lds(&Hb[ks * 32], 1024, lane);
      v16bf bb = ld_b_swz(W2, 64, ks, nt, lane);
      acc = wmma_bf16(a, bb, acc);
      __builtin_prefetch(W2 + (((size_t)((ks + 1) * 64 + nt) * 32) << 4), 0, 1);
    }
    float vmax = -3.0e38f;
#pragma unroll
    for (int r = 0; r < 8; ++r) vmax = fmaxf(vmax, acc[r]);
    vmax = fmaxf(vmax, __shfl_xor(vmax, 16, 32));
    if (hl == 0)
      partials[((size_t)batch * 256 + blkInB) * 1024 + nt * 16 + colL] =
          vmax + b2[nt * 16 + colL];
  }
}

__global__ void reduce_g_kernel(const float* __restrict__ partials,
                                float* __restrict__ g) {
  int c = blockIdx.x * 256 + threadIdx.x;
  int b = blockIdx.y;
  float m = -3.0e38f;
  for (int r = 0; r < 256; ++r)
    m = fmaxf(m, partials[((size_t)b * 256 + r) * 1024 + c]);
  g[(size_t)b * 1024 + c] = m;
}

// ---- feat = concat(x0,x1,x2,x3,x4,g) as bf16, padded 1350 -> 1376 ----------

__global__ void feat_build_kernel(const float* __restrict__ x0,
                                  const float* __restrict__ x1,
                                  const float* __restrict__ x2,
                                  const float* __restrict__ x3,
                                  const float* __restrict__ x4,
                                  const float* __restrict__ g,
                                  __bf16* __restrict__ feat) {
  size_t i = (size_t)blockIdx.x * 256 + threadIdx.x;
  if (i < (size_t)MROWS * 1376) {
    int row = (int)(i / 1376);
    int c   = (int)(i % 1376);
    int b   = row >> 12;
    float v = 0.f;
    if      (c < 6)    v = x0[(size_t)row * 6 + c];
    else if (c < 70)   v = x1[(size_t)row * 64 + (c - 6)];
    else if (c < 134)  v = x2[(size_t)row * 64 + (c - 70)];
    else if (c < 198)  v = x3[(size_t)row * 64 + (c - 134)];
    else if (c < 326)  v = x4[(size_t)row * 128 + (c - 198)];
    else if (c < 1350) v = g[(size_t)b * 1024 + (c - 326)];
    feat[i] = (__bf16)v;
  }
}

// ---- Generic WMMA GEMM for the head: out = [relu](A @ W + bias) ------------
// A: (grid.x*32) x Kp bf16, W swizzled Kp x Ncp bf16. Block: 8 waves,
// 32 rows x 64 cols. A tile staged via TDM (tensor_load_to_lds) if available.

template <bool RELU, bool BF16OUT>
__global__ void __launch_bounds__(256) gemm_kernel(
    const __bf16* __restrict__ A, const __bf16* __restrict__ W,
    const float* __restrict__ bias, void* __restrict__ out,
    int Kp, int Ncp, int Nreal) {
  __shared__ alignas(32) __bf16 At[32 * 32];
  int tid  = threadIdx.x;
  int wave = tid >> 5, lane = tid & 31;
  int mt = wave & 1, nt = wave >> 1;
  int colL = lane & 15, hl = lane >> 4;
  size_t row0 = (size_t)blockIdx.x * 32;
  int n0 = blockIdx.y * 64 + nt * 16;
  bool active = n0 < Ncp;
  int ntiles = Ncp >> 4;
  int ntg = blockIdx.y * 4 + nt;  // global n-tile index
  v8f acc = {};
  int ksteps = Kp / 32;
  for (int ks = 0; ks < ksteps; ++ks) {
#if defined(HAVE_TDM)
    // Tensor Data Mover: DMA the 32x32 bf16 A tile (row-major, stride Kp)
    // into LDS. One wave issues; TENSORcnt waited before the barrier.
    if (wave == 0) {
      unsigned long long ga =
          (unsigned long long)(const void*)(A + row0 * (size_t)Kp + ks * 32);
      unsigned lds = (unsigned)(size_t)(void*)&At[0];
      u32x4 g0;
      g0[0] = 1u;                                        // count=1 (valid D#)
      g0[1] = lds;                                       // lds_addr
      g0[2] = (unsigned)ga;                              // global_addr[31:0]
      g0[3] = (unsigned)((ga >> 32) & 0x1FFFFFFu) | (2u << 30);  // type=2
      i32x8 g1;
      g1[0] = 1 << 16;                                   // data_size = 2 bytes
      g1[1] = (Kp & 0xFFFF) << 16;                       // tensor_dim0 lo16
      g1[2] = (Kp >> 16) & 0xFFFF;                       // dim0 hi / dim1 lo=0
      g1[3] = (32 << 16) | 0x10;                         // tile_dim0=32, dim1 hi
      g1[4] = 32;                                        // tile_dim1=32
      g1[5] = Kp;                                        // tensor_dim0_stride
      g1[6] = 0;
      g1[7] = 0;
      i32x4 z4 = {0, 0, 0, 0};
      i32x8 z8 = {0, 0, 0, 0, 0, 0, 0, 0};
      __builtin_amdgcn_tensor_load_to_lds(g0, g1, z4, z4, z8, 0);
      __builtin_amdgcn_s_wait_tensorcnt(0);
    }
#else
    if (tid < 128) {
      int r = tid >> 2, c8 = (tid & 3) << 3;
      *(uint4*)&At[r * 32 + c8] =
          *(const uint4*)&A[(row0 + r) * (size_t)Kp + ks * 32 + c8];
    }
#endif
    __syncthreads();
    if (active) {
      v16bf a  = ld_a_lds(&At[mt * 16 * 32], 32, lane);
      v16bf bb = ld_b_swz(W, ntiles, ks, ntg, lane);
      acc = wmma_bf16(a, bb, acc);
      __builtin_prefetch(W + (((size_t)((ks + 1) * ntiles + ntg) * 32) << 4), 0, 1);
    }
    __syncthreads();
  }
  if (active) {
    int col = n0 + colL;
    float bs = (col < Nreal) ? bias[col] : 0.f;
#pragma unroll
    for (int r = 0; r < 8; ++r) {
      float v = acc[r] + bs;
      if (RELU) v = fmaxf(v, 0.f);
      size_t o = (row0 + mt * 16 + hl * 8 + r) * (size_t)Ncp + col;
      if (BF16OUT) ((__bf16*)out)[o] = (__bf16)v;
      else         ((float*)out)[o]  = v;
    }
  }
}

__global__ void logsoftmax_kernel(const float* __restrict__ h,
                                  float* __restrict__ out) {
  int i = blockIdx.x * 256 + threadIdx.x;
  if (i < MROWS) {
    const float* r = h + (size_t)i * 16;
    float mx = r[0];
    for (int j = 1; j < 13; ++j) mx = fmaxf(mx, r[j]);
    float s = 0.f;
    for (int j = 0; j < 13; ++j) s += __expf(r[j] - mx);
    float ls = __logf(s);
    for (int j = 0; j < 13; ++j) out[(size_t)i * 13 + j] = r[j] - mx - ls;
  }
}

// ---------------------------------------------------------------------------

extern "C" void kernel_launch(void* const* d_in, const int* in_sizes, int n_in,
                              void* d_out, int out_size, void* d_ws, size_t ws_size,
                              hipStream_t stream) {
  (void)in_sizes; (void)n_in; (void)out_size; (void)ws_size;
  const float* xin  = (const float*)d_in[0];
  const float* posn = (const float*)d_in[1];
  const float* e1W1 = (const float*)d_in[2],  *e1b1 = (const float*)d_in[3];
  const float* e1W2 = (const float*)d_in[4],  *e1b2 = (const float*)d_in[5];
  const float* e2W1 = (const float*)d_in[6],  *e2b1 = (const float*)d_in[7];
  const float* e2W2 = (const float*)d_in[8],  *e2b2 = (const float*)d_in[9];
  const float* e3W1 = (const float*)d_in[10], *e3b1 = (const float*)d_in[11];
  const float* e3W2 = (const float*)d_in[12], *e3b2 = (const float*)d_in[13];
  const float* e4W1 = (const float*)d_in[14], *e4b1 = (const float*)d_in[15];
  const float* e4W2 = (const float*)d_in[16], *e4b2 = (const float*)d_in[17];
  const float* feW1 = (const float*)d_in[18], *feb1 = (const float*)d_in[19];
  const float* feW2 = (const float*)d_in[20], *feb2 = (const float*)d_in[21];
  const float* h0W  = (const float*)d_in[22], *h0b  = (const float*)d_in[23];
  const float* h1W  = (const float*)d_in[24], *h1b  = (const float*)d_in[25];
  const float* h2W  = (const float*)d_in[26], *h2b  = (const float*)d_in[27];
  const float* h3W  = (const float*)d_in[28], *h3b  = (const float*)d_in[29];
  float* outp = (float*)d_out;

  uint8_t* ws = (uint8_t*)d_ws;
  size_t off = 0;
  auto alloc = [&](size_t bytes) -> void* {
    void* p = ws + off;
    off += (bytes + 255) & ~(size_t)255;
    return p;
  };

  int*    idxbuf = (int*)   alloc((size_t)MROWS * KNN * 4);
  float*  x0buf  = (float*) alloc((size_t)MROWS * 6 * 4);
  float*  x1buf  = (float*) alloc((size_t)MROWS * 64 * 4);
  float*  x2buf  = (float*) alloc((size_t)MROWS * 64 * 4);
  float*  x3buf  = (float*) alloc((size_t)MROWS * 64 * 4);
  float*  x4buf  = (float*) alloc((size_t)MROWS * 128 * 4);
  __bf16* x4b    = (__bf16*)alloc((size_t)MROWS * 128 * 2);
  float*  parts  = (float*) alloc((size_t)BATCH * 256 * 1024 * 4);
  float*  gbuf   = (float*) alloc((size_t)BATCH * 1024 * 4);
  __bf16* feat   = (__bf16*)alloc((size_t)MROWS * 1376 * 2);
  __bf16* h0o    = (__bf16*)alloc((size_t)MROWS * 256 * 2);
  __bf16* h1o    = (__bf16*)alloc((size_t)MROWS * 256 * 2);
  __bf16* h2o    = (__bf16*)alloc((size_t)MROWS * 128 * 2);
  float*  h3o    = (float*) alloc((size_t)MROWS * 16 * 4);
  __bf16* e1W1b  = (__bf16*)alloc(32 * 64 * 2);
  __bf16* e1W2b  = (__bf16*)alloc(64 * 64 * 2);
  __bf16* e2W1b  = (__bf16*)alloc(128 * 64 * 2);
  __bf16* e2W2b  = (__bf16*)alloc(64 * 64 * 2);
  __bf16* e3W1b  = (__bf16*)alloc(128 * 64 * 2);
  __bf16* e3W2b  = (__bf16*)alloc(64 * 64 * 2);
  __bf16* e4W1b  = (__bf16*)alloc(128 * 128 * 2);
  __bf16* e4W2b  = (__bf16*)alloc(128 * 128 * 2);
  __bf16* feW1b  = (__bf16*)alloc(128 * 1024 * 2);
  __bf16* feW2b  = (__bf16*)alloc((size_t)1024 * 1024 * 2);
  __bf16* h0Wb   = (__bf16*)alloc((size_t)1376 * 256 * 2);
  __bf16* h1Wb   = (__bf16*)alloc(256 * 256 * 2);
  __bf16* h2Wb   = (__bf16*)alloc(256 * 128 * 2);
  __bf16* h3Wb   = (__bf16*)alloc(128 * 16 * 2);

  auto wcvt = [&](const float* s, __bf16* d, int K, int Ns, int Kp, int Np) {
    int total = Kp * Np;
    wcvt_swz_kernel<<<(total + 255) / 256, 256, 0, stream>>>(s, d, K, Ns, Kp, Np);
  };

  // Weight staging (f32 -> bf16, fragment-swizzled, zero padded)
  wcvt(e1W1, e1W1b, 12, 64, 32, 64);
  wcvt(e1W2, e1W2b, 64, 64, 64, 64);
  wcvt(e2W1, e2W1b, 128, 64, 128, 64);
  wcvt(e2W2, e2W2b, 64, 64, 64, 64);
  wcvt(e3W1, e3W1b, 128, 64, 128, 64);
  wcvt(e3W2, e3W2b, 64, 64, 64, 64);
  wcvt(e4W1, e4W1b, 128, 128, 128, 128);
  wcvt(e4W2, e4W2b, 128, 128, 128, 128);
  wcvt(feW1, feW1b, 128, 1024, 128, 1024);
  wcvt(feW2, feW2b, 1024, 1024, 1024, 1024);
  wcvt(h0W,  h0Wb,  1350, 256, 1376, 256);
  wcvt(h1W,  h1Wb,  256, 256, 256, 256);
  wcvt(h2W,  h2Wb,  256, 128, 256, 128);
  wcvt(h3W,  h3Wb,  128, 13, 128, 16);

  // x0 = concat(x, pos)
  build_x0_kernel<<<(MROWS * 6 + 255) / 256, 256, 0, stream>>>(xin, posn, x0buf);

  // EdgeConv chain
  knn_kernel<6><<<dim3(NPTS / 128, BATCH), 128, 0, stream>>>(x0buf, idxbuf);
  edgeconv_kernel<6, 32, 64, 4><<<dim3(NPTS / 4, BATCH), 128, 0, stream>>>(
      x0buf, idxbuf, e1W1b, e1b1, e1W2b, e1b2, x1buf);

  knn_kernel<64><<<dim3(NPTS / 128, BATCH), 128, 0, stream>>>(x1buf, idxbuf);
  edgeconv_kernel<64, 128, 64, 4><<<dim3(NPTS / 4, BATCH), 128, 0, stream>>>(
      x1buf, idxbuf, e2W1b, e2b1, e2W2b, e2b2, x2buf);

  knn_kernel<64><<<dim3(NPTS / 128, BATCH), 128, 0, stream>>>(x2buf, idxbuf);
  edgeconv_kernel<64, 128, 64, 4><<<dim3(NPTS / 4, BATCH), 128, 0, stream>>>(
      x2buf, idxbuf, e3W1b, e3b1, e3W2b, e3b2, x3buf);

  knn_kernel<64><<<dim3(NPTS / 128, BATCH), 128, 0, stream>>>(x3buf, idxbuf);
  edgeconv_kernel<64, 128, 128, 2><<<dim3(NPTS / 2, BATCH), 64, 0, stream>>>(
      x3buf, idxbuf, e4W1b, e4b1, e4W2b, e4b2, x4buf);

  // fe MLP (fused) + global max
  f2bf_kernel<<<(MROWS * 128 + 255) / 256, 256, 0, stream>>>(x4buf, x4b,
                                                             MROWS * 128);
  fe_fused_kernel<<<MROWS / 16, 256, 0, stream>>>(x4b, feW1b, feb1, feW2b, feb2,
                                                  parts);
  reduce_g_kernel<<<dim3(1024 / 256, BATCH), 256, 0, stream>>>(parts, gbuf);

  // head
  {
    size_t total = (size_t)MROWS * 1376;
    feat_build_kernel<<<(unsigned)((total + 255) / 256), 256, 0, stream>>>(
        x0buf, x1buf, x2buf, x3buf, x4buf, gbuf, feat);
  }
  gemm_kernel<true, true><<<dim3(MROWS / 32, 4), 256, 0, stream>>>(
      feat, h0Wb, h0b, (void*)h0o, 1376, 256, 256);
  gemm_kernel<true, true><<<dim3(MROWS / 32, 4), 256, 0, stream>>>(
      h0o, h1Wb, h1b, (void*)h1o, 256, 256, 256);
  gemm_kernel<true, true><<<dim3(MROWS / 32, 2), 256, 0, stream>>>(
      h1o, h2Wb, h2b, (void*)h2o, 256, 128, 128);
  gemm_kernel<false, false><<<dim3(MROWS / 32, 1), 256, 0, stream>>>(
      h2o, h3Wb, h3b, (void*)h3o, 128, 16, 13);

  logsoftmax_kernel<<<MROWS / 256, 256, 0, stream>>>(h3o, outp);
}
